// Decoder_3298534884262
// MI455X (gfx1250) — matrix operand order repaired and checked by reading, not compiled, not run on hardware
//
#include <hip/hip_runtime.h>
#include <hip/hip_bf16.h>

#define Bc 64
#define Lc 128
#define Hc 1024
#define Ec 512
#define Vc 50257

typedef __bf16 bf16_t;
typedef __attribute__((ext_vector_type(16))) __bf16 v16bf;
typedef __attribute__((ext_vector_type(8)))  float  v8f;

#define WMMA_BF16(a, b, c)                                                    \
  __builtin_amdgcn_wmma_f32_16x16x32_bf16(false, (a), false, (b), (short)0,   \
                                          (c), false, false)

// ---------------------------------------------------------------------------
// WMMA operand layouts (CDNA5 ISA §7.12.2, wave32):
// A 16x32 bf16:  m = lane&15;  half=lane>>4:
//   e0..15 -> K = k0 + 16*half + e (contiguous order used by this code)
// B 32x16 bf16:  n = lane&15, element e -> K = k0 + 16*half + e (this code's order, same as A).
// D 16x16 f32:   n = lane&15, row m = 8*half + vgpr_j.
// ---------------------------------------------------------------------------
__device__ __forceinline__ v16bf cvt16(float4 x0, float4 x1, float4 x2,
                                       float4 x3) {
  v16bf r;
  r[0] = (bf16_t)x0.x;  r[1] = (bf16_t)x0.y;  r[2] = (bf16_t)x0.z;  r[3] = (bf16_t)x0.w;
  r[4] = (bf16_t)x1.x;  r[5] = (bf16_t)x1.y;  r[6] = (bf16_t)x1.z;  r[7] = (bf16_t)x1.w;
  r[8] = (bf16_t)x2.x;  r[9] = (bf16_t)x2.y;  r[10] = (bf16_t)x2.z; r[11] = (bf16_t)x2.w;
  r[12] = (bf16_t)x3.x; r[13] = (bf16_t)x3.y; r[14] = (bf16_t)x3.z; r[15] = (bf16_t)x3.w;
  return r;
}

// ---------------------------------------------------------------------------
// Pack B [K,N] fp32 (row-major) into bf16 WMMA B-operand tiles:
// layout [kt][nt][lane][16 bf16] ; tile linear index = kt*(N/16)+nt.
// One thread = one (tile, lane) slot -> 16 strided loads, 2x b128 store.
// ---------------------------------------------------------------------------
__global__ void pack_b_nn(const float* __restrict__ B, uint4* __restrict__ out,
                          int K, int N) {
  const int gid = blockIdx.x * 256 + threadIdx.x;
  const int lane = gid & 31;
  const int tile = gid >> 5;
  const int ntiles = N / 16;
  const int ktiles = K / 32;
  if (tile >= ktiles * ntiles) return;
  const int kt = tile / ntiles;
  const int nt = tile - kt * ntiles;
  const int lr = lane & 15;
  const int half = (lane >> 4) & 1;
  const float* p = B + (size_t)(kt * 32 + 16 * half) * N + nt * 16 + lr;
  union { uint4 u[2]; v16bf v; } bu;
#pragma unroll
  for (int e = 0; e < 16; ++e) bu.v[e] = (bf16_t)p[(size_t)e * N];
  uint4* dst = out + ((size_t)tile * 32 + lane) * 2;
  dst[0] = bu.u[0];
  dst[1] = bu.u[1];
}

// ---------------------------------------------------------------------------
// Software-pipelined NN core: acc[4] += A[16 rows x K] @ Bpacked (4 n-tiles).
// A is fp32 (converted in-register); B is pre-packed bf16 tiles.
// ---------------------------------------------------------------------------
__device__ __forceinline__ void nn_load(const float* arow, const uint4* pb0,
                                        size_t bstep, int k0, int kt, int half,
                                        float4 ra[4], uint4 rb[4][2]) {
  const float* p = arow + k0 + 16 * half;
  const float* q = arow + k0 + 16 * half + 8;
  ra[0] = *(const float4*)p;
  ra[1] = *(const float4*)(p + 4);
  ra[2] = *(const float4*)q;
  ra[3] = *(const float4*)(q + 4);
  const uint4* bp = pb0 + (size_t)kt * bstep;
#pragma unroll
  for (int t = 0; t < 4; ++t) {
    rb[t][0] = bp[t * 64];  // consecutive n-tiles: stride 32 lanes * 2 uint4
    rb[t][1] = bp[t * 64 + 1];
  }
}

__device__ __forceinline__ void nn_compute(const float4 ra[4],
                                           const uint4 rb[4][2], v8f acc[4]) {
  v16bf a = cvt16(ra[0], ra[1], ra[2], ra[3]);
#pragma unroll
  for (int t = 0; t < 4; ++t) {
    union { uint4 u[2]; v16bf v; } bu;
    bu.u[0] = rb[t][0];
    bu.u[1] = rb[t][1];
    acc[t] = WMMA_BF16(a, bu.v, acc[t]);
  }
}

__device__ __forceinline__ void nn_core(const float* arow, const uint4* Bpk,
                                        int nt0, int ntiles, int K, int lane,
                                        v8f acc[4]) {
  const int half = (lane >> 4) & 1;
  const uint4* pb0 = Bpk + ((size_t)nt0 * 32 + lane) * 2;
  const size_t bstep = (size_t)ntiles * 64;
  float4 ra0[4], ra1[4];
  uint4 rb0[4][2], rb1[4][2];
  const int nk = K / 32;  // even for all uses (K = 1024)
  nn_load(arow, pb0, bstep, 0, 0, half, ra0, rb0);
  for (int kt = 0; kt < nk; kt += 2) {
    nn_load(arow, pb0, bstep, (kt + 1) * 32, kt + 1, half, ra1, rb1);
    nn_compute(ra0, rb0, acc);
    if (kt + 2 < nk)
      nn_load(arow, pb0, bstep, (kt + 2) * 32, kt + 2, half, ra0, rb0);
    nn_compute(ra1, rb1, acc);
  }
}

// ---------------------------------------------------------------------------
// C[M,N] = A[M,K] @ B (B pre-packed). N multiple of 512.
// 256 threads = 8 waves; wave owns a 16x64 tile (4 accumulators).
// ---------------------------------------------------------------------------
__global__ void gemm_nn_packed(const float* __restrict__ A,
                               const uint4* __restrict__ Bpk,
                               float* __restrict__ C, int M, int N, int K) {
  const int wave = threadIdx.x >> 5;
  const int lane = threadIdx.x & 31;
  const int m0 = blockIdx.x * 16;
  const int nt0 = (blockIdx.y * 8 + wave) * 4;
  if (m0 >= M) return;

  v8f acc[4] = {{}, {}, {}, {}};
  nn_core(A + (size_t)(m0 + (lane & 15)) * K, Bpk, nt0, N / 16, K, lane, acc);

  const int lr = lane & 15;
  const int half = (lane >> 4) & 1;
#pragma unroll
  for (int t = 0; t < 4; ++t) {
    const int n = (nt0 + t) * 16 + lr;
#pragma unroll
    for (int j = 0; j < 8; ++j)
      C[(size_t)(m0 + half * 8 + j) * N + n] = acc[t][j];
  }
}

// ---------------------------------------------------------------------------
// Fused attention scores: score[bl] += sum_n tanh((ctx@U)[bl,n] + ah[b,n])*v[n]
// ctx@U intermediate never materialized; 4 sub-tiles summed locally, one
// atomicAdd per row per wave.
// ---------------------------------------------------------------------------
__global__ void gemm_nn_score(const float* __restrict__ A,     // ctx [B*L, K]
                              const uint4* __restrict__ Bpk,   // packed U
                              const float* __restrict__ ah,    // [B, N]
                              const float* __restrict__ vvec,  // [N]
                              float* __restrict__ score,       // [B*L]
                              int M, int N, int K, int Lctx) {
  const int wave = threadIdx.x >> 5;
  const int lane = threadIdx.x & 31;
  const int m0 = blockIdx.x * 16;
  const int nt0 = (blockIdx.y * 8 + wave) * 4;
  if (m0 >= M) return;

  v8f acc[4] = {{}, {}, {}, {}};
  nn_core(A + (size_t)(m0 + (lane & 15)) * K, Bpk, nt0, N / 16, K, lane, acc);

  const int lr = lane & 15;
  const int half = (lane >> 4) & 1;
  const int row0 = m0 + half * 8;  // first bl row this half covers
  const int bidx = row0 / Lctx;    // L=128 > 16 -> one b per tile

  float ptot[8] = {0.f, 0.f, 0.f, 0.f, 0.f, 0.f, 0.f, 0.f};
#pragma unroll
  for (int t = 0; t < 4; ++t) {
    const int n = (nt0 + t) * 16 + lr;
    const float vn = vvec[n];
    const float ahv = ah[(size_t)bidx * N + n];
#pragma unroll
    for (int j = 0; j < 8; ++j) ptot[j] += tanhf(acc[t][j] + ahv) * vn;
  }
#pragma unroll
  for (int j = 0; j < 8; ++j) {
    float p = ptot[j];
#pragma unroll
    for (int off = 1; off < 16; off <<= 1) p += __shfl_xor(p, off, 32);
    if (lr == 0) atomicAdd(&score[row0 + j], p);
  }
}

// ---------------------------------------------------------------------------
// C[M,N] = A[M,K] @ Bt[N,K]^T + bias (one-use fp32 weights, in-reg convert).
// All 20 raw b128 loads of a K-step issued into distinct regs before convert
// so loadcnt drains incrementally instead of blocking per-load.
// ---------------------------------------------------------------------------
__global__ void gemm_nt_bias(const float* __restrict__ A,
                             const float* __restrict__ Bt,
                             const float* __restrict__ bias,
                             float* __restrict__ C, int M, int N, int K) {
  const int wave = threadIdx.x >> 5;
  const int lane = threadIdx.x & 31;
  const int m0 = blockIdx.x * 16;
  const int n0 = (blockIdx.y * 8 + wave) * 64;
  if (m0 >= M || n0 >= N) return;  // wave-uniform: EXEC all-1 for WMMA

  const int lr = lane & 15;
  const int half = (lane >> 4) & 1;
  const float* arow = A + (size_t)(m0 + lr) * K;
  const float* brow[4];
#pragma unroll
  for (int t = 0; t < 4; ++t) {
    int n = n0 + 16 * t + lr;
    if (n >= N) n = N - 1;  // clamp; store guarded
    brow[t] = Bt + (size_t)n * K + 16 * half;
  }

  v8f acc[4] = {{}, {}, {}, {}};
  for (int k0 = 0; k0 < K; k0 += 32) {
    if (k0 + 32 < K) {
      __builtin_prefetch(arow + k0 + 32, 0, 0);
      __builtin_prefetch(brow[0] + k0 + 32, 0, 0);
    }
    float4 ra0 = *(const float4*)(arow + k0 + 16 * half);
    float4 ra1 = *(const float4*)(arow + k0 + 16 * half + 4);
    float4 ra2 = *(const float4*)(arow + k0 + 16 * half + 8);
    float4 ra3 = *(const float4*)(arow + k0 + 16 * half + 12);
    float4 rb[4][4];
#pragma unroll
    for (int t = 0; t < 4; ++t)
#pragma unroll
      for (int c = 0; c < 4; ++c)
        rb[t][c] = *(const float4*)(brow[t] + k0 + 4 * c);

    v16bf a = cvt16(ra0, ra1, ra2, ra3);
#pragma unroll
    for (int t = 0; t < 4; ++t) {
      v16bf b = cvt16(rb[t][0], rb[t][1], rb[t][2], rb[t][3]);
      acc[t] = WMMA_BF16(a, b, acc[t]);
    }
  }
#pragma unroll
  for (int t = 0; t < 4; ++t) {
    const int n = n0 + 16 * t + lr;
    if (n < N) {
      const float bv = bias ? bias[n] : 0.0f;
#pragma unroll
      for (int j = 0; j < 8; ++j)
        C[(size_t)(m0 + half * 8 + j) * N + n] = acc[t][j] + bv;
    }
  }
}

// ---------------------------------------------------------------------------
// Per-batch softmax over L (with pad mask) + weighted context sum -> [B,H]
// ---------------------------------------------------------------------------
__global__ void attn_finish(const float* __restrict__ score,
                            const unsigned char* __restrict__ pad,
                            const float* __restrict__ ctx,
                            float* __restrict__ out) {
  const int b = blockIdx.x;
  const int tid = threadIdx.x;
  __shared__ float w[Lc];
  __shared__ float red[256];

  float s = -1e30f;
  if (tid < Lc) {
    float sc = score[b * Lc + tid];
    if (pad[b * Lc + tid]) sc = -100000.0f;
    w[tid] = sc;
    s = sc;
  }
  red[tid] = s;
  __syncthreads();
  for (int st = 128; st > 0; st >>= 1) {
    if (tid < st) red[tid] = fmaxf(red[tid], red[tid + st]);
    __syncthreads();
  }
  const float mx = red[0];
  __syncthreads();
  float e = 0.0f;
  if (tid < Lc) {
    e = __expf(w[tid] - mx);
    w[tid] = e;
  }
  red[tid] = e;
  __syncthreads();
  for (int st = 128; st > 0; st >>= 1) {
    if (tid < st) red[tid] += red[tid + st];
    __syncthreads();
  }
  const float inv = 1.0f / red[0];
  __syncthreads();
  if (tid < Lc) w[tid] *= inv;
  __syncthreads();

  for (int h = tid; h < Hc; h += blockDim.x) {
    float acc = 0.0f;
    for (int l = 0; l < Lc; ++l)
      acc += w[l] * ctx[((size_t)b * Lc + l) * Hc + h];
    out[(size_t)b * Hc + h] = acc;
  }
}

// ---------------------------------------------------------------------------
// Merge gate: g = sigmoid(sum_h tanh(scl+sh)*wS - sum_h tanh(srl+sh)*wS)
// c_t = g*ac + (1-g)*ar ; builds GRU input x = [input | c_t]
// ---------------------------------------------------------------------------
__global__ void merge_kernel(const float* __restrict__ scl,
                             const float* __restrict__ srl,
                             const float* __restrict__ sh,
                             const float* __restrict__ wSw,
                             const float* __restrict__ ac,
                             const float* __restrict__ ar,
                             const float* __restrict__ input,
                             float* __restrict__ x) {
  const int b = blockIdx.x;
  const int tid = threadIdx.x;
  __shared__ float redc[256], redr[256];
  __shared__ float gshare;
  float pc = 0.0f, pr = 0.0f;
  for (int h = tid; h < Hc; h += 256) {
    const float shv = sh[b * Hc + h];
    const float wv = wSw[h];
    pc += tanhf(scl[b * Hc + h] + shv) * wv;
    pr += tanhf(srl[b * Hc + h] + shv) * wv;
  }
  redc[tid] = pc;
  redr[tid] = pr;
  __syncthreads();
  for (int st = 128; st > 0; st >>= 1) {
    if (tid < st) {
      redc[tid] += redc[tid + st];
      redr[tid] += redr[tid + st];
    }
    __syncthreads();
  }
  if (tid == 0) gshare = 1.0f / (1.0f + __expf(-(redc[0] - redr[0])));
  __syncthreads();
  const float g = gshare;
  for (int i = tid; i < Ec; i += 256)
    x[(size_t)b * (Ec + Hc) + i] = input[(size_t)b * Ec + i];
  for (int h = tid; h < Hc; h += 256) {
    const float a1 = ac[b * Hc + h], a2 = ar[b * Hc + h];
    x[(size_t)b * (Ec + Hc) + Ec + h] = g * a1 + (1.0f - g) * a2;
  }
}

// ---------------------------------------------------------------------------
// GRU cell (torch gate order r,z,n)
// ---------------------------------------------------------------------------
__global__ void gru_kernel(const float* __restrict__ gi,
                           const float* __restrict__ gh,
                           const float* __restrict__ hprev,
                           float* __restrict__ hout) {
  const int b = blockIdx.x;
  for (int h = threadIdx.x; h < Hc; h += 256) {
    const float ir = gi[b * 3 * Hc + h];
    const float iz = gi[b * 3 * Hc + Hc + h];
    const float in = gi[b * 3 * Hc + 2 * Hc + h];
    const float hr = gh[b * 3 * Hc + h];
    const float hz = gh[b * 3 * Hc + Hc + h];
    const float hn = gh[b * 3 * Hc + 2 * Hc + h];
    const float r = 1.0f / (1.0f + __expf(-(ir + hr)));
    const float z = 1.0f / (1.0f + __expf(-(iz + hz)));
    const float n = tanhf(in + r * hn);
    hout[b * Hc + h] = (1.0f - z) * n + z * hprev[b * Hc + h];
  }
}

// ---------------------------------------------------------------------------
// In-place row log-softmax over V
// ---------------------------------------------------------------------------
__global__ void logsoftmax_kernel(float* __restrict__ logits, int Vn) {
  const int b = blockIdx.x;
  const int tid = threadIdx.x;
  __shared__ float red[256];
  float mx = -1e30f;
  for (int v = tid; v < Vn; v += 256)
    mx = fmaxf(mx, logits[(size_t)b * Vn + v]);
  red[tid] = mx;
  __syncthreads();
  for (int st = 128; st > 0; st >>= 1) {
    if (tid < st) red[tid] = fmaxf(red[tid], red[tid + st]);
    __syncthreads();
  }
  mx = red[0];
  __syncthreads();
  float sum = 0.0f;
  for (int v = tid; v < Vn; v += 256)
    sum += __expf(logits[(size_t)b * Vn + v] - mx);
  red[tid] = sum;
  __syncthreads();
  for (int st = 128; st > 0; st >>= 1) {
    if (tid < st) red[tid] += red[tid + st];
    __syncthreads();
  }
  const float lse = mx + logf(red[0]);
  __syncthreads();
  for (int v = tid; v < Vn; v += 256) logits[(size_t)b * Vn + v] -= lse;
}

// ---------------------------------------------------------------------------
extern "C" void kernel_launch(void* const* d_in, const int* in_sizes, int n_in,
                              void* d_out, int out_size, void* d_ws,
                              size_t ws_size, hipStream_t stream) {
  const float* input  = (const float*)d_in[0];   // [B,1,E]
  const float* hidden = (const float*)d_in[1];   // [1,B,H] -> [B,H]
  const float* ctx_c  = (const float*)d_in[2];   // [B,L,H]
  const float* ctx_r  = (const float*)d_in[3];   // [B,L,H]
  const unsigned char* pad = (const unsigned char*)d_in[4];  // [B,L] bool
  const float* W      = (const float*)d_in[5];   // [H,H]
  const float* U      = (const float*)d_in[6];   // [H,H]
  const float* v      = (const float*)d_in[7];   // [H,1]
  const float* WSh_w  = (const float*)d_in[8];
  const float* WSh_b  = (const float*)d_in[9];
  const float* WSc_w  = (const float*)d_in[10];
  const float* WSc_b  = (const float*)d_in[11];
  const float* WSr_w  = (const float*)d_in[12];
  const float* WSr_b  = (const float*)d_in[13];
  const float* wS_w   = (const float*)d_in[14];
  const float* W_ih   = (const float*)d_in[16];  // [3H, H+E]
  const float* W_hh   = (const float*)d_in[17];  // [3H, H]
  const float* b_ih   = (const float*)d_in[18];
  const float* b_hh   = (const float*)d_in[19];
  const float* W_out  = (const float*)d_in[20];  // [V, H]
  const float* b_out  = (const float*)d_in[21];

  float* ws = (float*)d_ws;
  float* ah     = ws;                      // B*H
  float* scoreC = ah + (size_t)Bc * Hc;    // B*L
  float* scoreR = scoreC + Bc * Lc;        // B*L
  float* ac     = scoreR + Bc * Lc;        // B*H
  float* ar     = ac + (size_t)Bc * Hc;    // B*H
  float* sh     = ar + (size_t)Bc * Hc;    // B*H
  float* scl    = sh + (size_t)Bc * Hc;    // B*H
  float* srl    = scl + (size_t)Bc * Hc;   // B*H
  float* xbuf   = srl + (size_t)Bc * Hc;   // B*(E+H)
  float* gi     = xbuf + (size_t)Bc * (Ec + Hc);  // B*3H
  float* gh     = gi + (size_t)Bc * 3 * Hc;       // B*3H
  float* fend   = gh + (size_t)Bc * 3 * Hc;       // byte offset % 16 == 0
  uint4* Upk    = (uint4*)fend;                   // H*H bf16 = 131072 uint4
  uint4* Wpk    = Upk + (size_t)Hc * Hc / 8;

  float* logits = (float*)d_out;                    // [B,V]
  float* h_out  = (float*)d_out + (size_t)Bc * Vc;  // [B,H]

  // scores are accumulated with atomics -> zero them each call
  hipMemsetAsync(scoreC, 0, 2 * Bc * Lc * sizeof(float), stream);

  const dim3 blk(256);
  const int btiles = (Hc / 32) * (Hc / 16);  // 2048 tiles for HxH

  // pack reused NN-side weights into bf16 WMMA tiles
  pack_b_nn<<<dim3(btiles * 32 / 256), blk, 0, stream>>>(U, Upk, Hc, Hc);
  pack_b_nn<<<dim3(btiles * 32 / 256), blk, 0, stream>>>(W, Wpk, Hc, Hc);

  // ah = hidden @ W                       [64,1024]x[1024,1024]
  gemm_nn_packed<<<dim3(Bc / 16, Hc / 512), blk, 0, stream>>>(hidden, Wpk, ah,
                                                              Bc, Hc, Hc);

  // fused tanh(ah + ctx@U) . v -> scores  [8192,1024]x[1024,1024] each
  gemm_nn_score<<<dim3(Bc * Lc / 16, Hc / 512), blk, 0, stream>>>(
      ctx_c, Upk, ah, v, scoreC, Bc * Lc, Hc, Hc, Lc);
  gemm_nn_score<<<dim3(Bc * Lc / 16, Hc / 512), blk, 0, stream>>>(
      ctx_r, Upk, ah, v, scoreR, Bc * Lc, Hc, Hc, Lc);

  // softmax + weighted context sum
  attn_finish<<<Bc, 256, 0, stream>>>(scoreC, pad, ctx_c, ac);
  attn_finish<<<Bc, 256, 0, stream>>>(scoreR, pad, ctx_r, ar);

  // merge-gate linears (A @ W^T + b)
  gemm_nt_bias<<<dim3(Bc / 16, Hc / 512), blk, 0, stream>>>(hidden, WSh_w, WSh_b, sh, Bc, Hc, Hc);
  gemm_nt_bias<<<dim3(Bc / 16, Hc / 512), blk, 0, stream>>>(ac, WSc_w, WSc_b, scl, Bc, Hc, Hc);
  gemm_nt_bias<<<dim3(Bc / 16, Hc / 512), blk, 0, stream>>>(ar, WSr_w, WSr_b, srl, Bc, Hc, Hc);

  // gate + c_t + build x = [input | c_t]
  merge_kernel<<<Bc, 256, 0, stream>>>(scl, srl, sh, wS_w, ac, ar, input, xbuf);

  // GRU GEMMs
  gemm_nt_bias<<<dim3(Bc / 16, (3 * Hc) / 512), blk, 0, stream>>>(
      xbuf, W_ih, b_ih, gi, Bc, 3 * Hc, Ec + Hc);
  gemm_nt_bias<<<dim3(Bc / 16, (3 * Hc) / 512), blk, 0, stream>>>(
      hidden, W_hh, b_hh, gh, Bc, 3 * Hc, Hc);

  // GRU cell -> h_new (second output, written directly into d_out)
  gru_kernel<<<Bc, 256, 0, stream>>>(gi, gh, hidden, h_out);

  // logits = h_new @ W_out^T + b_out      [64,50257]
  gemm_nt_bias<<<dim3(Bc / 16, (Vc + 511) / 512), blk, 0, stream>>>(
      h_out, W_out, b_out, logits, Bc, Vc, Hc);

  // in-place log-softmax over V
  logsoftmax_kernel<<<Bc, 256, 0, stream>>>(logits, Vc);
}
